// LocalAttention_50551765074508
// MI455X (gfx1250) — matrix operand order repaired and checked by reading, not compiled, not run on hardware
//
#include <hip/hip_runtime.h>
#include <math.h>

typedef __attribute__((ext_vector_type(2)))  float    v2f;
typedef __attribute__((ext_vector_type(4)))  float    v4f;
typedef __attribute__((ext_vector_type(8)))  float    v8f;
typedef __attribute__((ext_vector_type(4)))  _Float16 v4h;
typedef __attribute__((ext_vector_type(8)))  _Float16 v8h;
typedef __attribute__((ext_vector_type(16))) _Float16 v16h;

#define WINDOW 128
#define EDIM   64
#define TSEQ   4096
#define NWIN   (TSEQ / WINDOW)   // 32
#define JDIM   (2 * WINDOW)      // 256 keys visible per window

// padded row strides (in halves) chosen for conflict-free, 16B-aligned frag loads
#define QROW  72                  // 144 B/row: lane stride 36 dw -> 16 distinct banks
#define KROW  72
#define VTROW 264                 // 528 B/row: lane stride 132 dw -> 16 distinct banks
#define PROW  40                  // 80 B/row:  lane stride 20 dw -> 16 distinct banks

// LDS layout in halves
#define SQ_OFF  0
#define SK_OFF  (SQ_OFF + WINDOW * QROW)            // 9216
#define SVT_OFF (SK_OFF + JDIM * KROW)              // 27648
#define SP_OFF  (SVT_OFF + EDIM * VTROW)            // 44544
#define SMEM_HALVES (SP_OFF + 8 * 16 * PROW)        // 49664 halves = 99328 B

static __device__ __forceinline__ v16h cat8(v8h lo, v8h hi) {
    return __builtin_shufflevector(lo, hi, 0, 1, 2, 3, 4, 5, 6, 7,
                                           8, 9, 10, 11, 12, 13, 14, 15);
}

__global__ __launch_bounds__(256)
void local_attn_f16_wmma(const float* __restrict__ q,
                         const float* __restrict__ k,
                         const float* __restrict__ v,
                         float* __restrict__ out)
{
    extern __shared__ _Float16 smem[];
    _Float16* sQ  = smem + SQ_OFF;   // [128][QROW]  Q * scale, row-major
    _Float16* sK  = smem + SK_OFF;   // [256][KROW]  K rows, row-major
    _Float16* sVt = smem + SVT_OFF;  // [64][VTROW]  V transposed (e-major)
    _Float16* sP  = smem + SP_OFF;   // per-wave [16][PROW] P staging

    const int bh = blockIdx.x / NWIN;
    const int w  = blockIdx.x % NWIN;
    const size_t base = (size_t)bh * TSEQ * EDIM;
    const float* qg = q + base + (size_t)w * WINDOW * EDIM;
    const float* kg = k + base;
    const float* vg = v + base;
    float* og = out + base + (size_t)w * WINDOW * EDIM;

    const int tid = threadIdx.x;
    const float scale = 0.125f;  // 64^-0.5

    // ---- stage Q (scaled, f16): 2048 float4 ----
    for (int i = tid; i < (WINDOW * EDIM) / 4; i += 256) {
        int row = i >> 4, c4 = i & 15;
        v4f t4 = ((const v4f*)qg)[i] * scale;
        *(v4h*)(sQ + row * QROW + c4 * 4) = __builtin_convertvector(t4, v4h);
    }
    // ---- stage K (f16 row-major) and V (f16 transposed), zero-fill OOB ----
    for (int i = tid; i < (JDIM * EDIM) / 4; i += 256) {
        int row = i >> 4, c4 = i & 15;
        int gr  = w * WINDOW - WINDOW + row;
        v4f kk = {0.0f, 0.0f, 0.0f, 0.0f};
        v4f vv = {0.0f, 0.0f, 0.0f, 0.0f};
        if (gr >= 0) {
            kk = ((const v4f*)(kg + (size_t)gr * EDIM))[c4];
            vv = ((const v4f*)(vg + (size_t)gr * EDIM))[c4];
        }
        *(v4h*)(sK + row * KROW + c4 * 4) = __builtin_convertvector(kk, v4h);
        v4h vh = __builtin_convertvector(vv, v4h);
        int e0 = c4 * 4;
        sVt[(e0 + 0) * VTROW + row] = vh.x;
        sVt[(e0 + 1) * VTROW + row] = vh.y;
        sVt[(e0 + 2) * VTROW + row] = vh.z;
        sVt[(e0 + 3) * VTROW + row] = vh.w;
    }
    __syncthreads();

    const int wave = tid >> 5;     // 0..7, owns query rows [wave*16, wave*16+16)
    const int lane = tid & 31;
    const int l16  = lane & 15;
    const int half = lane >> 4;

    // ---------------- S = (Q*scale) @ K^T : 16 x 256, f32 accum ----------------
    // A frag (16x32 f16): elems 0-7 = K 8*half+0..7 ; elems 8-15 = K 16+8*half+0..7
    // B frag (32x16 f16): elems 0-7 = K 8*half+0..7 ; elems 8-15 = K 16+8*half+0..7 at column l16
    v8f S[16];
    const _Float16* aBase = sQ + (wave * 16 + l16) * QROW;
#pragma unroll
    for (int jt = 0; jt < 16; ++jt) {
        const _Float16* bBase = sK + (jt * 16 + l16) * KROW;
        v8f acc = {};
#pragma unroll
        for (int eb = 0; eb < 2; ++eb) {   // K-chunks of 32 over e=64
            v8h alo = *(const v8h*)(aBase + eb * 32 + half * 8);
            v8h ahi = *(const v8h*)(aBase + eb * 32 + 16 + half * 8);
            v8h blo = *(const v8h*)(bBase + eb * 32 + half * 8);
            v8h bhi = *(const v8h*)(bBase + eb * 32 + 16 + half * 8);
            acc = __builtin_amdgcn_wmma_f32_16x16x32_f16(false, cat8(alo, ahi),
                                                         false, cat8(blo, bhi),
                                                         (short)0, acc, false, false);
        }
        S[jt] = acc;
    }

    // ---------------- mask: pad (w==0, j<128) and causal (j > i+128) ----------------
    const float NEG = -3.4028235e38f;
#pragma unroll
    for (int jt = 0; jt < 16; ++jt) {
#pragma unroll
        for (int r = 0; r < 8; ++r) {
            int i = wave * 16 + r + half * 8;   // C/D layout row
            int j = jt * 16 + l16;              // C/D layout col
            bool pad    = (w == 0) && (j < WINDOW);
            bool causal = j > i + WINDOW;
            if (pad || causal) S[jt][r] = NEG;
        }
    }

    // ---------------- softmax over j (f32; rows live in 16-lane halves) ----------------
    float rowmax[8];
#pragma unroll
    for (int r = 0; r < 8; ++r) {
        float m = NEG;
#pragma unroll
        for (int jt = 0; jt < 16; ++jt) m = fmaxf(m, S[jt][r]);
#pragma unroll
        for (int d = 1; d < 16; d <<= 1) m = fmaxf(m, __shfl_xor(m, d, 32));
        rowmax[r] = m;
    }
#pragma unroll
    for (int r = 0; r < 8; ++r) {
        float sum = 0.0f;
#pragma unroll
        for (int jt = 0; jt < 16; ++jt) {
            float p = __expf(S[jt][r] - rowmax[r]);
            S[jt][r] = p;
            sum += p;
        }
#pragma unroll
        for (int d = 1; d < 16; d <<= 1) sum += __shfl_xor(sum, d, 32);
        float inv = 1.0f / sum;
#pragma unroll
        for (int jt = 0; jt < 16; ++jt) S[jt][r] *= inv;
    }

    // ---------------- O = P @ V : 16 x 64 ----------------
    v8f O[4] = {{}, {}, {}, {}};
    _Float16* myP = sP + wave * (16 * PROW);
#pragma unroll
    for (int c = 0; c < 8; ++c) {          // j-chunks of 32
        // C-layout f32 -> f16 A-staging tile [16][32] (rows padded to PROW)
#pragma unroll
        for (int t = 0; t < 2; ++t) {
            int jt = c * 2 + t;
#pragma unroll
            for (int r = 0; r < 8; ++r)
                myP[(r + half * 8) * PROW + t * 16 + l16] = (_Float16)S[jt][r];
        }
        __syncthreads();
        v8h plo = *(const v8h*)(myP + l16 * PROW + half * 8);
        v8h phi = *(const v8h*)(myP + l16 * PROW + 16 + half * 8);
        v16h aP = cat8(plo, phi);
#pragma unroll
        for (int et = 0; et < 4; ++et) {
            const _Float16* vb = sVt + (et * 16 + l16) * VTROW + c * 32 + half * 8;
            v8h blo = *(const v8h*)(vb);
            v8h bhi = *(const v8h*)(vb + 16);
            O[et] = __builtin_amdgcn_wmma_f32_16x16x32_f16(false, aP,
                                                           false, cat8(blo, bhi),
                                                           (short)0, O[et], false, false);
        }
        __syncthreads();
    }

    // ---------------- writeback (C/D layout -> row-major global, f32) ----------------
#pragma unroll
    for (int et = 0; et < 4; ++et) {
#pragma unroll
        for (int r = 0; r < 8; ++r) {
            int row = wave * 16 + r + half * 8;
            og[(size_t)row * EDIM + et * 16 + l16] = O[et][r];
        }
    }
}

extern "C" void kernel_launch(void* const* d_in, const int* in_sizes, int n_in,
                              void* d_out, int out_size, void* d_ws, size_t ws_size,
                              hipStream_t stream) {
    (void)in_sizes; (void)n_in; (void)out_size; (void)d_ws; (void)ws_size;
    const float* q = (const float*)d_in[0];
    const float* k = (const float*)d_in[1];
    const float* v = (const float*)d_in[2];
    float* out = (float*)d_out;

    const size_t smem_bytes = (size_t)SMEM_HALVES * sizeof(_Float16);  // ~97 KB
    (void)hipFuncSetAttribute((const void*)local_attn_f16_wmma,
                              hipFuncAttributeMaxDynamicSharedMemorySize,
                              (int)smem_bytes);

    const int n_blocks = 2 * 16 * NWIN;  // b*h*windows = 1024
    local_attn_f16_wmma<<<dim3(n_blocks), dim3(256), smem_bytes, stream>>>(q, k, v, out);
}
